// SpGraphAttentionLayer_83193516523656
// MI455X (gfx1250) — matrix operand order repaired and checked by reading, not compiled, not run on hardware
//
#include <hip/hip_runtime.h>
#include <hip/hip_bf16.h>
#include <math.h>

#define N_NODES 50000
#define DIN 512
#define DOUT 256
#define NEG_SLOPE 0.01f

typedef __attribute__((ext_vector_type(16))) __bf16 v16bf;
typedef __attribute__((ext_vector_type(8)))  __bf16 v8bf;
typedef __attribute__((ext_vector_type(8)))  float  v8f;
typedef __attribute__((ext_vector_type(4)))  unsigned int u32x4;
typedef __attribute__((ext_vector_type(8)))  int i32x8;
typedef __attribute__((ext_vector_type(4)))  int i32x4;

__device__ __forceinline__ __bf16 f32_to_bf16_rne(float f) {
    unsigned u = __builtin_bit_cast(unsigned, f);
    unsigned r = (u + 0x7FFFu + ((u >> 16) & 1u)) >> 16;
    return __builtin_bit_cast(__bf16, (unsigned short)r);
}

// Zero output accumulator + rowsum (harness poisons with 0xAA).
__global__ void k_zero(float* __restrict__ out, float* __restrict__ rowsum) {
    int tid = blockIdx.x * blockDim.x + threadIdx.x;   // N_NODES*DOUT threads
    out[tid] = 0.0f;
    if (tid < N_NODES) rowsum[tid] = 0.0f;
}

// Convert x (f32, row-major [N][512]) -> xb (bf16), 8 elements/thread.
__global__ void k_xbf(const float* __restrict__ x, __bf16* __restrict__ xb) {
    size_t tid = (size_t)blockIdx.x * blockDim.x + threadIdx.x;
    v8f v = *(const v8f*)(x + tid * 8);
    v8bf o;
    #pragma unroll
    for (int e = 0; e < 8; ++e) o[e] = f32_to_bf16_rne(v[e]);
    *(v8bf*)(xb + tid * 8) = o;
}

// Transpose + convert W[k][n] (f32) -> Wt[n][k] (bf16): B fragments contiguous.
__global__ void k_wt(const float* __restrict__ W, __bf16* __restrict__ Wt) {
    int tid = blockIdx.x * blockDim.x + threadIdx.x;   // DIN*DOUT threads
    int n = tid >> 9;
    int k = tid & (DIN - 1);
    Wt[tid] = f32_to_bf16_rne(W[(size_t)k * DOUT + n]);
}

// h = x @ W. One block per 16-row strip; 8 waves x two 16x16 N-tiles = 256 cols.
// A strip (16x512 bf16 = 16 KB) staged into LDS by the Tensor Data Mover.
__global__ void __launch_bounds__(256) k_gemm(const __bf16* __restrict__ xb,
                                              const __bf16* __restrict__ Wt,
                                              float* __restrict__ h) {
    __shared__ __bf16 As[16 * DIN];                    // 16 KB
    const int m0   = blockIdx.x * 16;
    const int wid  = threadIdx.x >> 5;
    const int lane = threadIdx.x & 31;
    const int hi   = lane >> 4;
    const int lr   = lane & 15;

    if (threadIdx.x < 32) {
        // TDM descriptor (CDNA5 ISA): 2D tile, 512x16
        // bf16 elements, row stride 512. LDS offset = low 32 bits of generic
        // pointer (LDS aperture truncation, ISA §10.2).
        unsigned lds_off = (unsigned)(unsigned long long)(&As[0]);
        unsigned long long ga =
            (unsigned long long)(const void*)(xb + (size_t)m0 * DIN);
        u32x4 g0;
        g0[0] = 1u;                                   // count=1, user mode
        g0[1] = lds_off;                              // lds_addr (bytes)
        g0[2] = (unsigned)(ga & 0xFFFFFFFFu);         // global_addr[31:0]
        g0[3] = (unsigned)((ga >> 32) & 0x01FFFFFFu)  // global_addr[56:32]
              | (2u << 30);                           // type=2 (image)
        i32x8 g1;
        g1[0] = (int)(1u << 16);                      // data_size=1 -> 2 bytes
        g1[1] = (int)(512u << 16);                    // tensor_dim0[15:0]
        g1[2] = (int)(16u << 16);                     // tensor_dim1[15:0]
        g1[3] = (int)(512u << 16);                    // tile_dim0
        g1[4] = 16;                                   // tile_dim1 (tile_dim2=0)
        g1[5] = 512;                                  // tensor_dim0_stride lo32
        g1[6] = 0;
        g1[7] = 0;
        i32x4 gz4 = {0, 0, 0, 0};
        i32x8 gz8 = {0, 0, 0, 0, 0, 0, 0, 0};
        // amdgpu-toolchain (clang-23) 6-arg form.
        __builtin_amdgcn_tensor_load_to_lds(g0, g1, gz4, gz4, gz8, 0);
        __builtin_amdgcn_s_wait_tensorcnt(0);
    }
    __syncthreads();

    const int n0 = wid * 32;                          // this wave: cols n0..n0+31
    const __bf16* w0 = Wt + (size_t)(n0 + lr) * DIN;  // B col for tile 0
    const __bf16* w1 = w0 + 16 * DIN;                 // B col for tile 1
    const __bf16* arow = As + lr * DIN;               // A row for this lane

    v8f acc0 = {}, acc1 = {};
    #pragma unroll 4
    for (int k0 = 0; k0 < DIN; k0 += 32) {
        const int ka = k0 + hi * 16;
        // A fragment: elems 0..15 = K+hi*16 .. K+hi*16+15 (contiguous).
        v8bf aLo = *(const v8bf*)(arow + ka);
        v8bf aHi = *(const v8bf*)(arow + ka + 8);
        v16bf A = __builtin_shufflevector(aLo, aHi,
                                          0, 1, 2, 3, 4, 5, 6, 7,
                                          8, 9, 10, 11, 12, 13, 14, 15);
        // B fragments: lane = column, 16 contiguous K at k0+hi*16.
        v16bf B0 = *(const v16bf*)(w0 + k0 + hi * 16);
        v16bf B1 = *(const v16bf*)(w1 + k0 + hi * 16);
        acc0 = __builtin_amdgcn_wmma_f32_16x16x32_bf16(false, A, false, B0,
                                                       (short)0, acc0, false, false);
        acc1 = __builtin_amdgcn_wmma_f32_16x16x32_bf16(false, A, false, B1,
                                                       (short)0, acc1, false, false);
    }
    // C/D layout: VGPR r holds row m0 + r + hi*8, column n0 + lr.
    float* hp0 = h + (size_t)m0 * DOUT + n0 + lr;
    float* hp1 = hp0 + 16;
    #pragma unroll
    for (int r = 0; r < 8; ++r) {
        hp0[(size_t)(r + hi * 8) * DOUT] = acc0[r];
        hp1[(size_t)(r + hi * 8) * DOUT] = acc1[r];
    }
}

// s_src[i] = h[i,:].a[0:256], s_dst[i] = h[i,:].a[256:512]. One wave per row.
__global__ void __launch_bounds__(256) k_scores(const float* __restrict__ h,
                                                const float* __restrict__ a,
                                                float* __restrict__ s_src,
                                                float* __restrict__ s_dst) {
    const int wid  = threadIdx.x >> 5;
    const int lane = threadIdx.x & 31;
    const int row  = blockIdx.x * 8 + wid;
    v8f hv = *(const v8f*)(h + (size_t)row * DOUT + lane * 8);
    v8f a1 = *(const v8f*)(a + lane * 8);
    v8f a2 = *(const v8f*)(a + DOUT + lane * 8);
    float ss = 0.f, sd = 0.f;
    #pragma unroll
    for (int e = 0; e < 8; ++e) { ss += hv[e] * a1[e]; sd += hv[e] * a2[e]; }
    #pragma unroll
    for (int off = 16; off > 0; off >>= 1) {
        ss += __shfl_down(ss, off, 32);
        sd += __shfl_down(sd, off, 32);
    }
    if (lane == 0) { s_src[row] = ss; s_dst[row] = sd; }
}

// Per-edge: e = exp(-leaky_relu(s_src[src] + s_dst[dst])); rowsum[src] += e.
__global__ void k_edges(const int* __restrict__ esrc, const int* __restrict__ edst,
                        const float* __restrict__ s_src, const float* __restrict__ s_dst,
                        float* __restrict__ e_val, float* __restrict__ rowsum, int E) {
    int tid = blockIdx.x * blockDim.x + threadIdx.x;
    if (tid >= E) return;
    int s = esrc[tid], d = edst[tid];
    float sc = s_src[s] + s_dst[d];
    float lr = sc > 0.f ? sc : NEG_SLOPE * sc;
    float ev = expf(-lr);
    e_val[tid] = ev;
    __hip_atomic_fetch_add(rowsum + s, ev, __ATOMIC_RELAXED, __HIP_MEMORY_SCOPE_AGENT);
}

// Numerator SpMM: out[src,:] += e * h[dst,:]. One wave per edge, 8 cols/lane.
// h (51.2 MB) stays resident in the 192 MB L2, so gathers + atomics are on-chip.
__global__ void __launch_bounds__(256) k_spmm(const int* __restrict__ esrc,
                                              const int* __restrict__ edst,
                                              const float* __restrict__ e_val,
                                              const float* __restrict__ h,
                                              float* __restrict__ out, int E) {
    const int wid  = threadIdx.x >> 5;
    const int lane = threadIdx.x & 31;
    const int edge = blockIdx.x * 8 + wid;
    if (edge >= E) return;
    const int s = esrc[edge], d = edst[edge];
    const float ev = e_val[edge];
    v8f hv = *(const v8f*)(h + (size_t)d * DOUT + lane * 8);
    float* op = out + (size_t)s * DOUT + lane * 8;
    #pragma unroll
    for (int e = 0; e < 8; ++e)
        __hip_atomic_fetch_add(op + e, ev * hv[e],
                               __ATOMIC_RELAXED, __HIP_MEMORY_SCOPE_AGENT);
}

// out = elu(out / rowsum[row]) in place.
__global__ void k_final(float* __restrict__ out, const float* __restrict__ rowsum) {
    int tid = blockIdx.x * blockDim.x + threadIdx.x;   // N_NODES*DOUT threads
    float v = out[tid] / rowsum[tid >> 8];             // DOUT == 256
    out[tid] = v > 0.f ? v : (expf(v) - 1.0f);
}

extern "C" void kernel_launch(void* const* d_in, const int* in_sizes, int n_in,
                              void* d_out, int out_size, void* d_ws, size_t ws_size,
                              hipStream_t stream) {
    const float* x    = (const float*)d_in[0];
    const float* W    = (const float*)d_in[1];
    const float* a    = (const float*)d_in[2];
    const int*   esrc = (const int*)d_in[3];
    const int*   edst = (const int*)d_in[4];
    const int    E    = in_sizes[3];
    float* out = (float*)d_out;

    // Workspace slabs (256 B aligned), total ~105.5 MB.
    char* ws = (char*)d_ws;
    auto align_up = [](size_t v) { return (v + 255u) & ~(size_t)255u; };
    __bf16* xb = (__bf16*)ws;  ws += align_up((size_t)N_NODES * DIN * sizeof(__bf16));
    __bf16* Wt = (__bf16*)ws;  ws += align_up((size_t)DIN * DOUT * sizeof(__bf16));
    float* h      = (float*)ws; ws += align_up((size_t)N_NODES * DOUT * sizeof(float));
    float* s_src  = (float*)ws; ws += align_up((size_t)N_NODES * sizeof(float));
    float* s_dst  = (float*)ws; ws += align_up((size_t)N_NODES * sizeof(float));
    float* e_val  = (float*)ws; ws += align_up((size_t)E * sizeof(float));
    float* rowsum = (float*)ws;

    k_zero  <<<N_NODES, 256, 0, stream>>>(out, rowsum);
    k_xbf   <<<(N_NODES * DIN) / (256 * 8), 256, 0, stream>>>(x, xb);
    k_wt    <<<(DIN * DOUT) / 256, 256, 0, stream>>>(W, Wt);
    k_gemm  <<<N_NODES / 16, 256, 0, stream>>>(xb, Wt, h);
    k_scores<<<N_NODES / 8, 256, 0, stream>>>(h, a, s_src, s_dst);
    k_edges <<<(E + 255) / 256, 256, 0, stream>>>(esrc, edst, s_src, s_dst, e_val, rowsum, E);
    k_spmm  <<<(E + 7) / 8, 256, 0, stream>>>(esrc, edst, e_val, h, out, E);
    k_final <<<(N_NODES * DOUT) / 256, 256, 0, stream>>>(out, rowsum);
}
